// LongformerForSentenceClassification_60318520705307
// MI455X (gfx1250) — matrix operand order repaired and checked by reading, not compiled, not run on hardware
//
#include <hip/hip_runtime.h>
#include <hip/hip_bf16.h>
#include <math.h>

typedef __attribute__((ext_vector_type(16))) _Float16 v16h;
typedef __attribute__((ext_vector_type(8)))  float    v8f;

#define SEP 2
#define B_  8
#define S_  4096
#define H_  768
#define MS_ 64

// ---------------------------------------------------------------------------
// Kernel 1: per-row segment prep. One block of 256 threads per batch row.
// Computes exclusive-cumsum segment ids of SEP flags, the reference's weight
// mask, per-segment counts -> reciprocal, and packs (seg | weight<<7) per token.
// ---------------------------------------------------------------------------
__global__ void pool_prep(const int* __restrict__ ids,
                          unsigned char* __restrict__ segw,
                          float* __restrict__ inv) {
  const int b = blockIdx.x;
  const int tid = threadIdx.x;
  __shared__ int s_scan[256];
  __shared__ int s_first;
  __shared__ unsigned short s_seg[S_];
  __shared__ float s_cnt[MS_ + 8];

  if (tid == 0) s_first = S_;
  for (int i = tid; i < MS_ + 1; i += 256) s_cnt[i] = 0.0f;
  __syncthreads();

  int carry = 0;
  for (int c = 0; c < S_ / 256; ++c) {
    const int t = c * 256 + tid;
    const int sep = (ids[b * S_ + t] == SEP) ? 1 : 0;
    s_scan[tid] = sep;
    __syncthreads();
    // Hillis-Steele inclusive scan over 256
    for (int off = 1; off < 256; off <<= 1) {
      int add = (tid >= off) ? s_scan[tid - off] : 0;
      __syncthreads();
      s_scan[tid] += add;
      __syncthreads();
    }
    const int incl = s_scan[tid];
    const int seg = carry + incl - sep;        // exclusive cumsum
    s_seg[t] = (unsigned short)seg;
    if (sep) atomicMin(&s_first, t);
    const int total = s_scan[255];             // chunk total (post-barrier)
    __syncthreads();                           // protect s_scan before reuse
    carry += total;
  }
  const int n_sep = carry;
  const int first = s_first;
  __syncthreads();

  for (int c = 0; c < S_ / 256; ++c) {
    const int t = c * 256 + tid;
    const int sep = (ids[b * S_ + t] == SEP) ? 1 : 0;
    const int seg = (int)s_seg[t];
    int w = sep ? (t == first ? 1 : 0) : 1;
    if (t == S_ - 1 && seg == n_sep && n_sep > 0) w = 0;
    const int segc = seg > MS_ ? MS_ : seg;
    if (w) atomicAdd(&s_cnt[segc], 1.0f);
    segw[b * S_ + t] = (unsigned char)(segc | (w << 7));
  }
  __syncthreads();
  if (tid < MS_) {
    const float cnt = s_cnt[tid];
    inv[b * MS_ + tid] = (cnt > 0.0f) ? (1.0f / cnt) : 0.0f;
  }
}

// ---------------------------------------------------------------------------
// Kernel 2: bandwidth kernel. Streams `hidden` (non-temporal: 100 MB one-shot
// stream must not pollute L2), run-length accumulates per segment, flushes
// with global f32 atomics into sums[b][seg][h] (65 segs).
// grid: (tchunk=16, hchunk=3, b=8), block 256. Each block: 256 tokens x 256 h.
// ---------------------------------------------------------------------------
__global__ void pool_sum(const float* __restrict__ hidden,
                         const unsigned char* __restrict__ segw,
                         float* __restrict__ sums) {
  const int b  = blockIdx.z;
  const int hc = blockIdx.y;
  const int tc = blockIdx.x;
  const int h  = hc * 256 + threadIdx.x;
  const int t0 = tc * 256;

  __shared__ unsigned char s_sw[256];
  s_sw[threadIdx.x] = segw[b * S_ + t0 + threadIdx.x];
  __syncthreads();

  const float* hp = hidden + (size_t)b * S_ * H_ + (size_t)t0 * H_ + h;
  float acc = 0.0f;
  int cur = s_sw[0] & 127;
  #pragma unroll 4
  for (int i = 0; i < 256; ++i) {
    const unsigned char sw = s_sw[i];
    const int s = sw & 127;
    if (s != cur) {
      if (acc != 0.0f) atomicAdd(&sums[((size_t)b * (MS_ + 1) + cur) * H_ + h], acc);
      acc = 0.0f;
      cur = s;
    }
    if (sw >> 7) acc += __builtin_nontemporal_load(hp + (size_t)i * H_);
  }
  atomicAdd(&sums[((size_t)b * (MS_ + 1) + cur) * H_ + h], acc);
}

// ---------------------------------------------------------------------------
// Kernel 3: finalize means -> f16 sent [512, 768]
// ---------------------------------------------------------------------------
__global__ void pool_finalize(const float* __restrict__ sums,
                              const float* __restrict__ inv,
                              _Float16* __restrict__ sent) {
  const int idx = blockIdx.x * 256 + threadIdx.x;   // < 8*64*768
  const int h = idx % H_;
  const int s = (idx / H_) % MS_;
  const int b = idx / (H_ * MS_);
  const float v = sums[((size_t)b * (MS_ + 1) + s) * H_ + h] * inv[b * MS_ + s];
  sent[idx] = (_Float16)v;
}

// ---------------------------------------------------------------------------
// Kernel 4: tiled transpose + f32->f16:  in[K,N] -> out[N,K]
// block dim3(32,8), grid dim3(N/32, K/32)
// ---------------------------------------------------------------------------
__global__ void transpose_f32_to_f16(const float* __restrict__ in,
                                     _Float16* __restrict__ out,
                                     int K, int N) {
  __shared__ float tile[32][33];
  const int k0 = blockIdx.y * 32;
  const int n0 = blockIdx.x * 32;
  const int tx = threadIdx.x, ty = threadIdx.y;
  #pragma unroll
  for (int r = 0; r < 32; r += 8)
    tile[ty + r][tx] = in[(size_t)(k0 + ty + r) * N + (n0 + tx)];
  __syncthreads();
  #pragma unroll
  for (int r = 0; r < 32; r += 8)
    out[(size_t)(n0 + ty + r) * K + (k0 + tx)] = (_Float16)tile[tx][ty + r];
}

// ---------------------------------------------------------------------------
// Fragment load helpers (ISA 7.12.2 layouts, 16-bit A/B, wave32)
// ---------------------------------------------------------------------------
union fragu { v16h h; unsigned int u[8]; };

__device__ __forceinline__ v16h load_a_frag(const _Float16* ap, int k0, int half) {
  fragu a;
  #pragma unroll
  for (int j = 0; j < 4; ++j) {
    a.u[j]     = *(const unsigned int*)(ap + k0 + half * 8 + 2 * j);
    a.u[j + 4] = *(const unsigned int*)(ap + k0 + 16 + half * 8 + 2 * j);
  }
  return a.h;
}

__device__ __forceinline__ v16h load_b_frag(const _Float16* bp, int k0, int half) {
  fragu b;
  #pragma unroll
  for (int j = 0; j < 4; ++j) {
    b.u[j]     = *(const unsigned int*)(bp + k0 + half * 8 + 2 * j);
    b.u[j + 4] = *(const unsigned int*)(bp + k0 + 16 + half * 8 + 2 * j);
  }
  return b.h;
}

__device__ __forceinline__ float gelu_exact(float v) {
  return 0.5f * v * (1.0f + erff(v * 0.70710678118654752f));
}

// ---------------------------------------------------------------------------
// Kernel 5a: WMMA GEMM + bias + exact GELU, f16 out. 32x32 macro-tile per
// wave (2x2 accumulators): 4 fragment loads -> 4 WMMAs per k-step, i.e.
// 2 b128 loads per v_wmma instead of 4 (operand reuse in registers).
// A: [M,K] f16 row-major. Bt: [N,K] f16. grid = (M/32)*(N/32)/8 blocks x 256.
// ---------------------------------------------------------------------------
__global__ void wmma_gemm_gelu_2x2(const _Float16* __restrict__ A,
                                   const _Float16* __restrict__ Bt,
                                   const float* __restrict__ bias,
                                   _Float16* __restrict__ out,
                                   int M, int N, int K) {
  const int wave = threadIdx.x >> 5;
  const int lane = threadIdx.x & 31;
  const int tile = blockIdx.x * 8 + wave;
  const int ntn  = N >> 5;                 // macro-tiles along N
  const int mt = (tile / ntn) << 5;        // macro-tile origin (rows)
  const int nt = (tile % ntn) << 5;        // macro-tile origin (cols)

  const int l16  = lane & 15;
  const int half = lane >> 4;

  const _Float16* ap0 = A  + (size_t)(mt + l16) * K;
  const _Float16* ap1 = A  + (size_t)(mt + 16 + l16) * K;
  const _Float16* bp0 = Bt + (size_t)(nt + l16) * K;
  const _Float16* bp1 = Bt + (size_t)(nt + 16 + l16) * K;

  v8f c00 = {0.f,0.f,0.f,0.f,0.f,0.f,0.f,0.f};
  v8f c01 = c00, c10 = c00, c11 = c00;

  for (int k0 = 0; k0 < K; k0 += 32) {
    const v16h a0 = load_a_frag(ap0, k0, half);
    const v16h a1 = load_a_frag(ap1, k0, half);
    const v16h b0 = load_b_frag(bp0, k0, half);
    const v16h b1 = load_b_frag(bp1, k0, half);
    c00 = __builtin_amdgcn_wmma_f32_16x16x32_f16(false, a0, false, b0, (short)0, c00, false, false);
    c01 = __builtin_amdgcn_wmma_f32_16x16x32_f16(false, a0, false, b1, (short)0, c01, false, false);
    c10 = __builtin_amdgcn_wmma_f32_16x16x32_f16(false, a1, false, b0, (short)0, c10, false, false);
    c11 = __builtin_amdgcn_wmma_f32_16x16x32_f16(false, a1, false, b1, (short)0, c11, false, false);
  }

  // C/D layout: VGPR r -> rows r (lanes 0-15) and r+8 (lanes 16-31), col = lane&15
  const float bv0 = bias[nt + l16];
  const float bv1 = bias[nt + 16 + l16];
  #pragma unroll
  for (int r = 0; r < 8; ++r) {
    const int row0 = mt + half * 8 + r;
    const int row1 = row0 + 16;
    out[(size_t)row0 * N + nt + l16]      = (_Float16)gelu_exact(c00[r] + bv0);
    out[(size_t)row0 * N + nt + 16 + l16] = (_Float16)gelu_exact(c01[r] + bv1);
    out[(size_t)row1 * N + nt + l16]      = (_Float16)gelu_exact(c10[r] + bv0);
    out[(size_t)row1 * N + nt + 16 + l16] = (_Float16)gelu_exact(c11[r] + bv1);
  }
}

// ---------------------------------------------------------------------------
// Kernel 5b: 1x1-tile WMMA GEMM (for the parallelism-limited GEMM2: small
// M*N, long K -> keep 512 independent waves instead of 128).
// ---------------------------------------------------------------------------
__global__ void wmma_gemm_gelu_1x1(const _Float16* __restrict__ A,
                                   const _Float16* __restrict__ Bt,
                                   const float* __restrict__ bias,
                                   _Float16* __restrict__ out,
                                   int M, int N, int K) {
  const int wave = threadIdx.x >> 5;
  const int lane = threadIdx.x & 31;
  const int tile = blockIdx.x * 8 + wave;
  const int ntn  = N >> 4;
  const int mt = (tile / ntn) << 4;
  const int nt = (tile % ntn) << 4;

  const int l16  = lane & 15;
  const int half = lane >> 4;

  const _Float16* ap = A  + (size_t)(mt + l16) * K;
  const _Float16* bp = Bt + (size_t)(nt + l16) * K;

  v8f c = {0.f,0.f,0.f,0.f,0.f,0.f,0.f,0.f};
  for (int k0 = 0; k0 < K; k0 += 32) {
    const v16h a = load_a_frag(ap, k0, half);
    const v16h b = load_b_frag(bp, k0, half);
    c = __builtin_amdgcn_wmma_f32_16x16x32_f16(false, a, false, b, (short)0, c, false, false);
  }

  const int ncol = nt + l16;
  const float bv = bias[ncol];
  #pragma unroll
  for (int r = 0; r < 8; ++r) {
    const int row = mt + half * 8 + r;
    out[(size_t)row * N + ncol] = (_Float16)gelu_exact(c[r] + bv);
  }
}

// ---------------------------------------------------------------------------
// Kernel 6: final 256 -> 2 head (tiny), f32 logits out [8,64,2]
// ---------------------------------------------------------------------------
__global__ void head_final(const _Float16* __restrict__ x2,
                           const float* __restrict__ W3,
                           const float* __restrict__ b3,
                           float* __restrict__ out) {
  const int m = blockIdx.x * 256 + threadIdx.x;  // < 512
  float a0 = b3[0], a1 = b3[1];
  #pragma unroll 4
  for (int k = 0; k < 256; ++k) {
    const float x = (float)x2[m * 256 + k];
    a0 += x * W3[k * 2 + 0];
    a1 += x * W3[k * 2 + 1];
  }
  out[m * 2 + 0] = a0;
  out[m * 2 + 1] = a1;
}

extern "C" void kernel_launch(void* const* d_in, const int* in_sizes, int n_in,
                              void* d_out, int out_size, void* d_ws, size_t ws_size,
                              hipStream_t stream) {
  (void)in_sizes; (void)n_in; (void)out_size; (void)ws_size;
  const float* hidden = (const float*)d_in[0];
  const int*   ids    = (const int*)d_in[1];
  const float* W1     = (const float*)d_in[2];
  const float* b1     = (const float*)d_in[3];
  const float* W2     = (const float*)d_in[4];
  const float* b2     = (const float*)d_in[5];
  const float* W3     = (const float*)d_in[6];
  const float* b3     = (const float*)d_in[7];
  float* out = (float*)d_out;

  char* ws = (char*)d_ws;
  size_t off = 0;
  auto carve = [&](size_t bytes) -> char* {
    char* p = ws + off;
    off = (off + bytes + 255) & ~(size_t)255;
    return p;
  };
  float*         sums = (float*)carve((size_t)B_ * (MS_ + 1) * H_ * 4);
  unsigned char* segw = (unsigned char*)carve((size_t)B_ * S_);
  float*         inv  = (float*)carve((size_t)B_ * MS_ * 4);
  _Float16*      sent = (_Float16*)carve((size_t)B_ * MS_ * H_ * 2);
  _Float16*      W1t  = (_Float16*)carve((size_t)4096 * H_ * 2);
  _Float16*      W2t  = (_Float16*)carve((size_t)256 * 4096 * 2);
  _Float16*      x1   = (_Float16*)carve((size_t)512 * 4096 * 2);
  _Float16*      x2   = (_Float16*)carve((size_t)512 * 256 * 2);

  hipMemsetAsync(sums, 0, (size_t)B_ * (MS_ + 1) * H_ * 4, stream);

  pool_prep<<<B_, 256, 0, stream>>>(ids, segw, inv);
  pool_sum<<<dim3(S_ / 256, H_ / 256, B_), 256, 0, stream>>>(hidden, segw, sums);
  pool_finalize<<<(B_ * MS_ * H_) / 256, 256, 0, stream>>>(sums, inv, sent);

  transpose_f32_to_f16<<<dim3(4096 / 32, H_ / 32), dim3(32, 8), 0, stream>>>(W1, W1t, H_, 4096);
  transpose_f32_to_f16<<<dim3(256 / 32, 4096 / 32), dim3(32, 8), 0, stream>>>(W2, W2t, 4096, 256);

  // GEMM1: M=512, N=4096, K=768 -> (16*128)=2048 macro-tiles / 8 waves = 256 blocks
  wmma_gemm_gelu_2x2<<<256, 256, 0, stream>>>(sent, W1t, b1, x1, 512, 4096, 768);
  // GEMM2: M=512, N=256, K=4096 -> 512 tiles / 8 waves = 64 blocks (parallelism-limited)
  wmma_gemm_gelu_1x1<<<64, 256, 0, stream>>>(x1, W2t, b2, x2, 512, 256, 4096);

  head_final<<<2, 256, 0, stream>>>(x2, W3, b3, out);
}
